// PerformerAttentionSoftmax_79465484911161
// MI455X (gfx1250) — matrix operand order repaired and checked by reading, not compiled, not run on hardware
//
#include <hip/hip_runtime.h>
#include <hip/hip_bf16.h>

typedef __bf16 bf16_t;
typedef __attribute__((ext_vector_type(16))) __bf16 v16bf;
typedef __attribute__((ext_vector_type(8)))  __bf16 v8bf;
typedef __attribute__((ext_vector_type(8)))  float  v8f;
typedef __attribute__((ext_vector_type(4)))  unsigned int u32x4;
typedef __attribute__((ext_vector_type(4)))  int i32x4;
typedef __attribute__((ext_vector_type(8)))  int i32x8;

#define EMBED 768
#define HEADS 8
#define HDIM  96
#define DAUG  112          // HDIM + 16 pad (col 96 = ones -> k_sum via GEMM)
#define NRF   256
#define BB    8
#define NN    4096
#define BH    (BB*HEADS)   // 64
#define ROWS  (BB*NN)      // 32768
#define BHN   (BH*NN)      // 262144

// ---- WMMA bf16 16x16x32 operand index helpers (CDNA5 ISA 7.12.2) ----
// A (16x32, 16-bit): lane&15 = M; dword j holds K pair:
//   j<4 -> K = 8*half + 2j ; j>=4 -> K = 16 + 8*half + 2*(j&3)
__device__ __forceinline__ int a_k(int j, int half) {
  return ((j < 4) ? 0 : 16) + 8 * half + 2 * (j & 3);
}
// B (32x16, 16-bit): lane&15 = N; dword j holds K pair, same order as A:
//   j<4 -> K = 8*half + 2j ; j>=4 -> K = 16 + 8*half + 2*(j&3)
__device__ __forceinline__ int b_k(int j, int half) {
  return ((j < 4) ? 0 : 16) + 8 * half + 2 * (j & 3);
}
__device__ __forceinline__ v8f wmma_bf16(v16bf a, v16bf b, v8f c) {
  return __builtin_amdgcn_wmma_f32_16x16x32_bf16(false, a, false, b, (short)0, c,
                                                 false, false);
}

// ---- Stage a [rows x rowlen] f32 tile (row stride `stride` elems) into LDS
// via the Tensor Data Mover (CDNA5 ISA ch.8 D# descriptor). One wave issues
// (TDM ignores EXEC, so guard to a single wave), waits TENSORcnt, then block
// barrier publishes LDS to all 8 waves.
__device__ __forceinline__ void stage_tile_f32(const float* __restrict__ src,
                                               float* __restrict__ lds,
                                               int rows, int rowlen, int stride) {
#if __has_builtin(__builtin_amdgcn_tensor_load_to_lds)
  if (threadIdx.x < 32) {
    unsigned long long ga = (unsigned long long)src;
    u32x4 g0;
    g0[0] = 1u;                                          // count=1, user mode
    g0[1] = (unsigned)(unsigned long long)lds;           // LDS byte address
    g0[2] = (unsigned)(ga & 0xffffffffull);              // global_addr[31:0]
    g0[3] = (unsigned)((ga >> 32) & 0x1ffffffull) | (2u << 30);  // [56:32] | type=2
    i32x8 g1;
    g1[0] = (int)(2u << 16);                             // data_size = 4 bytes
    g1[1] = (int)(((unsigned)rowlen & 0xffffu) << 16);   // tensor_dim0[15:0]
    g1[2] = (int)((((unsigned)rowlen >> 16) & 0xffffu) |
                  (((unsigned)rows & 0xffffu) << 16));   // dim0 hi | tensor_dim1 lo
    g1[3] = (int)((((unsigned)rows >> 16) & 0xffffu) |
                  (((unsigned)rowlen & 0xffffu) << 16)); // dim1 hi | tile_dim0
    g1[4] = (int)((unsigned)rows & 0xffffu);             // tile_dim1 | tile_dim2=0
    g1[5] = (int)stride;                                 // tensor_dim0_stride[31:0]
    g1[6] = 0;
    g1[7] = 0;
    i32x4 z4 = {0, 0, 0, 0};
#if __clang_major__ >= 23
    i32x8 z8 = {0, 0, 0, 0, 0, 0, 0, 0};
    __builtin_amdgcn_tensor_load_to_lds(g0, g1, z4, z4, z8, 0);
#else
    __builtin_amdgcn_tensor_load_to_lds(g0, g1, z4, z4, 0);
#endif
    __builtin_amdgcn_s_wait_tensorcnt(0);
  }
  __syncthreads();
#else
  for (int i = threadIdx.x; i < rows * rowlen; i += (int)blockDim.x)
    lds[i] = src[(i / rowlen) * stride + (i % rowlen)];
  __syncthreads();
#endif
}

// ---------------- Kernel 1: QKV GEMM  out = x @ qkv_w^T + b ----------------
// Block = 8 waves = 128 token rows x 16 out cols; B panel (16x768 f32, 48KB)
// staged once per block via TDM and reused by all 8 waves.
__global__ __launch_bounds__(256) void qkv_wmma_kernel(
    const float* __restrict__ x, const float* __restrict__ w,
    const float* __restrict__ bias, bf16_t* __restrict__ q_s,
    bf16_t* __restrict__ k_s, bf16_t* __restrict__ v_aug) {
  __shared__ float Bs[16 * EMBED];
  const int NT = (3 * EMBED) / 16;  // 144
  const int nt = blockIdx.x % NT;
  const int mb = blockIdx.x / NT;
  stage_tile_f32(w + (size_t)(nt * 16) * EMBED, Bs, 16, EMBED, EMBED);

  const int wave = threadIdx.x >> 5;
  const int lane = threadIdx.x & 31, half = lane >> 4, l15 = lane & 15;
  const int mt = mb * 8 + wave;
  const int m = mt * 16 + l15;
  const int n = nt * 16 + l15;
  const float2* xrow2 = (const float2*)(x + (size_t)m * EMBED);
  const float2* brow2 = (const float2*)(Bs + l15 * EMBED);
  v8f acc = {};
  for (int kk = 0; kk < EMBED; kk += 32) {
    v16bf a, b;
#pragma unroll
    for (int j = 0; j < 8; ++j) {
      float2 fa = xrow2[(kk + a_k(j, half)) >> 1];
      a[2 * j] = (bf16_t)fa.x;
      a[2 * j + 1] = (bf16_t)fa.y;
      float2 fb = brow2[(kk + b_k(j, half)) >> 1];
      b[2 * j] = (bf16_t)fb.x;
      b[2 * j + 1] = (bf16_t)fb.y;
    }
    acc = wmma_bf16(a, b, acc);
  }
  const float bv = bias[n];
  const float scale = 0.10206207261596577f;  // 96^-0.5
#pragma unroll
  for (int r = 0; r < 8; ++r) {
    int row = mt * 16 + r + 8 * half;  // global token row
    int btok = row >> 12;              // /4096
    int ntok = row & 4095;
    float val = acc[r] + bv;
    if (n < EMBED) {
      int h = n / HDIM, d = n % HDIM;
      q_s[(size_t)((btok * HEADS + h) * NN + ntok) * HDIM + d] =
          (bf16_t)(val * scale);
    } else if (n < 2 * EMBED) {
      int c = n - EMBED, h = c / HDIM, d = c % HDIM;
      k_s[(size_t)((btok * HEADS + h) * NN + ntok) * HDIM + d] =
          (bf16_t)(val * scale);
    } else {
      int c = n - 2 * EMBED, h = c / HDIM, d = c % HDIM;
      v_aug[(size_t)((btok * HEADS + h) * NN + ntok) * DAUG + d] = (bf16_t)val;
    }
  }
}

// -------- Kernel 2: 0.5*||q||^2, 0.5*||k||^2 and ones/zero pad of v_aug -----
__global__ __launch_bounds__(256) void norm_ones_kernel(
    const bf16_t* __restrict__ q_s, const bf16_t* __restrict__ k_s,
    float* __restrict__ q_norm, float* __restrict__ k_norm,
    bf16_t* __restrict__ v_aug) {
  int idx = blockIdx.x * blockDim.x + threadIdx.x;
  if (idx >= BHN) return;
  const bf16_t* q = q_s + (size_t)idx * HDIM;
  const bf16_t* k = k_s + (size_t)idx * HDIM;
  float sq = 0.f, sk = 0.f;
  for (int d = 0; d < HDIM; ++d) {
    float qa = (float)q[d], kb = (float)k[d];
    sq += qa * qa;
    sk += kb * kb;
  }
  q_norm[idx] = 0.5f * sq;
  k_norm[idx] = 0.5f * sk;
  bf16_t* vr = v_aug + (size_t)idx * DAUG;
  vr[HDIM] = (bf16_t)1.0f;
  for (int d = HDIM + 1; d < DAUG; ++d) vr[d] = (bf16_t)0.0f;
}

// --------- Kernel 3: phi = exp(qk @ rf^T - norm)  (run for q and k) ---------
// transpose=0: row-major [bh*N, 256] (phi_q, consumed by den/attn GEMM as A).
// transpose=1: [bh, 256, N] packed v8bf stores (phi_k, A of the kv GEMM ->
//              contiguous dword A loads there).
__global__ __launch_bounds__(256) void phi_wmma_kernel(
    const bf16_t* __restrict__ qk, const float* __restrict__ rf,
    const float* __restrict__ norm, bf16_t* __restrict__ phi, int transpose) {
  __shared__ float Rs[16 * HDIM];   // 6KB rf panel
  const int NT = NRF / 16;          // 16
  const int nt = blockIdx.x % NT;
  const int mb = blockIdx.x / NT;
  stage_tile_f32(rf + (size_t)(nt * 16) * HDIM, Rs, 16, HDIM, HDIM);

  const int wave = threadIdx.x >> 5;
  const int lane = threadIdx.x & 31, half = lane >> 4, l15 = lane & 15;
  const int mt = mb * 8 + wave;
  const int m = mt * 16 + l15;
  const int n = nt * 16 + l15;
  const unsigned* arow32 = (const unsigned*)(qk + (size_t)m * HDIM);
  const float2* brow2 = (const float2*)(Rs + l15 * HDIM);
  v8f acc = {};
  for (int kk = 0; kk < HDIM; kk += 32) {
    i32x8 ai;
    v16bf b;
#pragma unroll
    for (int j = 0; j < 8; ++j) {
      ai[j] = arow32[(kk + a_k(j, half)) >> 1];
      float2 fb = brow2[(kk + b_k(j, half)) >> 1];
      b[2 * j] = (bf16_t)fb.x;
      b[2 * j + 1] = (bf16_t)fb.y;
    }
    acc = wmma_bf16(__builtin_bit_cast(v16bf, ai), b, acc);
  }
  if (transpose) {
    int bh = (mt * 16) >> 12;
    int ntok0 = ((mt * 16) & 4095) + 8 * half;
    v8bf vals;
#pragma unroll
    for (int r = 0; r < 8; ++r) {
      int row = mt * 16 + r + 8 * half;
      vals[r] = (bf16_t)expf(acc[r] - norm[row]);
    }
    *(v8bf*)(phi + (size_t)(bh * NRF + n) * NN + ntok0) = vals;
  } else {
#pragma unroll
    for (int r = 0; r < 8; ++r) {
      int row = mt * 16 + r + 8 * half;
      phi[(size_t)row * NRF + n] = (bf16_t)expf(acc[r] - norm[row]);
    }
  }
}

// --------- Kernel 4: kv_aug[bh,256,112] = phi_k^T @ v_aug (K = 4096) --------
// A = phi_kT [bh,256,N] -> contiguous dword loads along K (tokens).
__global__ __launch_bounds__(256) void kv_wmma_kernel(
    const bf16_t* __restrict__ phi_kT, const bf16_t* __restrict__ v_aug,
    float* __restrict__ kv) {
  const int wave = (blockIdx.x * blockDim.x + threadIdx.x) >> 5;
  const int RT = NRF / 16;         // 16
  const int CT = DAUG / 16;        // 7
  const int total = BH * RT * CT;  // 7168
  if (wave >= total) return;
  int bh = wave / (RT * CT);
  int rem = wave % (RT * CT);
  int rt = rem / CT, ct = rem % CT;
  const int lane = threadIdx.x & 31, half = lane >> 4, l15 = lane & 15;
  const int mrf = rt * 16 + l15;  // A row = rf index
  const int n = ct * 16 + l15;    // B col = d index
  const unsigned* arow32 =
      (const unsigned*)(phi_kT + (size_t)(bh * NRF + mrf) * NN);
  const bf16_t* va = v_aug + (size_t)bh * NN * DAUG;
  v8f acc = {};
  for (int kk = 0; kk < NN; kk += 32) {
    i32x8 ai;
    v16bf b;
#pragma unroll
    for (int j = 0; j < 8; ++j) {
      ai[j] = arow32[(kk + a_k(j, half)) >> 1];
      int kb = kk + b_k(j, half);
      b[2 * j] = va[(size_t)kb * DAUG + n];
      b[2 * j + 1] = va[(size_t)(kb + 1) * DAUG + n];
    }
    acc = wmma_bf16(__builtin_bit_cast(v16bf, ai), b, acc);
  }
#pragma unroll
  for (int r = 0; r < 8; ++r) {
    int row = rt * 16 + r + 8 * half;
    kv[(size_t)bh * NRF * DAUG + (size_t)row * DAUG + n] = acc[r];
  }
}

// --------- Kernel 5: denominator = phi_q . kv[:,96] (= phi_q . k_sum) -------
__global__ __launch_bounds__(256) void den_kernel(
    const bf16_t* __restrict__ phi_q, const float* __restrict__ kv,
    float* __restrict__ den) {
  int idx = blockIdx.x * blockDim.x + threadIdx.x;
  if (idx >= BHN) return;
  int bh = idx >> 12;
  const bf16_t* pq = phi_q + (size_t)idx * NRF;
  const float* ks = kv + (size_t)bh * NRF * DAUG + HDIM;  // ones column
  float s = 0.f;
  for (int mm = 0; mm < NRF; ++mm) s += (float)pq[mm] * ks[(size_t)mm * DAUG];
  den[idx] = s;
}

// ------ Kernel 6: numerator = phi_q @ kv (K=256); attn = num/den; transpose -
__global__ __launch_bounds__(256) void attn_wmma_kernel(
    const bf16_t* __restrict__ phi_q, const float* __restrict__ kv,
    const float* __restrict__ den, bf16_t* __restrict__ attn) {
  const int wave = (blockIdx.x * blockDim.x + threadIdx.x) >> 5;
  const int CT = HDIM / 16;           // 6
  const int total = (BHN / 16) * CT;  // 98304
  if (wave >= total) return;
  const int mt = wave / CT, ct = wave % CT;
  const int lane = threadIdx.x & 31, half = lane >> 4, l15 = lane & 15;
  const int m = mt * 16 + l15;     // global (bh*N + ntok) row
  const int bh = (mt * 16) >> 12;  // constant within tile
  const int n = ct * 16 + l15;     // d in [0,96)
  const unsigned* arow32 = (const unsigned*)(phi_q + (size_t)m * NRF);
  const float* kvb = kv + (size_t)bh * NRF * DAUG;
  v8f acc = {};
  for (int kk = 0; kk < NRF; kk += 32) {
    i32x8 ai;
    v16bf b;
#pragma unroll
    for (int j = 0; j < 8; ++j) {
      ai[j] = arow32[(kk + a_k(j, half)) >> 1];
      int kb = kk + b_k(j, half);
      b[2 * j] = (bf16_t)kvb[(size_t)kb * DAUG + n];
      b[2 * j + 1] = (bf16_t)kvb[(size_t)(kb + 1) * DAUG + n];
    }
    acc = wmma_bf16(__builtin_bit_cast(v16bf, ai), b, acc);
  }
#pragma unroll
  for (int r = 0; r < 8; ++r) {
    int row = mt * 16 + r + 8 * half;
    int b_ = row >> 15;  // /(HEADS*NN)
    int h = (row >> 12) & 7;
    int ntok = row & 4095;
    float val = acc[r] / fmaxf(den[row], 1e-6f);
    attn[(size_t)(b_ * NN + ntok) * EMBED + h * HDIM + n] = (bf16_t)val;
  }
}

// ---------------- Kernel 7: out = attn @ proj_w^T + proj_b (f32) ------------
// Same block tiling as QKV: TDM-staged 16x768 B panel shared by 8 waves.
__global__ __launch_bounds__(256) void proj_wmma_kernel(
    const bf16_t* __restrict__ attn, const float* __restrict__ w,
    const float* __restrict__ bias, float* __restrict__ out) {
  __shared__ float Bs[16 * EMBED];
  const int NT = EMBED / 16;  // 48
  const int nt = blockIdx.x % NT;
  const int mb = blockIdx.x / NT;
  stage_tile_f32(w + (size_t)(nt * 16) * EMBED, Bs, 16, EMBED, EMBED);

  const int wave = threadIdx.x >> 5;
  const int lane = threadIdx.x & 31, half = lane >> 4, l15 = lane & 15;
  const int mt = mb * 8 + wave;
  const int m = mt * 16 + l15;
  const int n = nt * 16 + l15;
  const unsigned* arow32 = (const unsigned*)(attn + (size_t)m * EMBED);
  const float2* brow2 = (const float2*)(Bs + l15 * EMBED);
  v8f acc = {};
  for (int kk = 0; kk < EMBED; kk += 32) {
    i32x8 ai;
    v16bf b;
#pragma unroll
    for (int j = 0; j < 8; ++j) {
      ai[j] = arow32[(kk + a_k(j, half)) >> 1];
      float2 fb = brow2[(kk + b_k(j, half)) >> 1];
      b[2 * j] = (bf16_t)fb.x;
      b[2 * j + 1] = (bf16_t)fb.y;
    }
    acc = wmma_bf16(__builtin_bit_cast(v16bf, ai), b, acc);
  }
  float bv = bias[n];
#pragma unroll
  for (int r = 0; r < 8; ++r) {
    int row = mt * 16 + r + 8 * half;
    out[(size_t)row * EMBED + n] = acc[r] + bv;
  }
}

extern "C" void kernel_launch(void* const* d_in, const int* in_sizes, int n_in,
                              void* d_out, int out_size, void* d_ws,
                              size_t ws_size, hipStream_t stream) {
  (void)in_sizes; (void)n_in; (void)out_size; (void)ws_size;
  const float* x      = (const float*)d_in[0];
  const float* qkv_w  = (const float*)d_in[1];
  const float* qkv_b  = (const float*)d_in[2];
  const float* proj_w = (const float*)d_in[3];
  const float* proj_b = (const float*)d_in[4];
  const float* rf     = (const float*)d_in[5];
  float* out = (float*)d_out;

  char* ws = (char*)d_ws;
  size_t off = 0;
  auto carve = [&](size_t bytes) -> char* {
    char* p = ws + off;
    off += (bytes + 255) & ~(size_t)255;
    return p;
  };
  bf16_t* q_s    = (bf16_t*)carve((size_t)BHN * HDIM * 2);
  bf16_t* k_s    = (bf16_t*)carve((size_t)BHN * HDIM * 2);
  bf16_t* v_aug  = (bf16_t*)carve((size_t)BHN * DAUG * 2);
  float*  q_norm = (float*) carve((size_t)BHN * 4);
  float*  k_norm = (float*) carve((size_t)BHN * 4);
  bf16_t* phi_q  = (bf16_t*)carve((size_t)BHN * NRF * 2);
  bf16_t* phi_kT = (bf16_t*)carve((size_t)BHN * NRF * 2);
  float*  kv     = (float*) carve((size_t)BH * NRF * DAUG * 4);
  // Aliases, safe by stream ordering: q_norm/q_s are dead after phi kernels.
  float*  den  = q_norm;
  bf16_t* attn = q_s;

  dim3 blk(256);
  qkv_wmma_kernel<<<(ROWS / 128) * ((3 * EMBED) / 16), blk, 0, stream>>>(
      x, qkv_w, qkv_b, q_s, k_s, v_aug);
  norm_ones_kernel<<<(BHN + 255) / 256, blk, 0, stream>>>(q_s, k_s, q_norm,
                                                          k_norm, v_aug);
  phi_wmma_kernel<<<(BHN / 128) * (NRF / 16), blk, 0, stream>>>(q_s, rf, q_norm,
                                                                phi_q, 0);
  phi_wmma_kernel<<<(BHN / 128) * (NRF / 16), blk, 0, stream>>>(k_s, rf, k_norm,
                                                                phi_kT, 1);
  {
    int waves = BH * (NRF / 16) * (DAUG / 16);
    kv_wmma_kernel<<<(waves + 7) / 8, blk, 0, stream>>>(phi_kT, v_aug, kv);
  }
  den_kernel<<<(BHN + 255) / 256, blk, 0, stream>>>(phi_q, kv, den);
  {
    int waves = (BHN / 16) * (HDIM / 16);
    attn_wmma_kernel<<<(waves + 7) / 8, blk, 0, stream>>>(phi_q, kv, den, attn);
  }
  proj_wmma_kernel<<<(ROWS / 128) * (EMBED / 16), blk, 0, stream>>>(
      attn, proj_w, proj_b, out);
}
